// GraphAttentionLayer_1039382086143
// MI455X (gfx1250) — matrix operand order repaired and checked by reading, not compiled, not run on hardware
//
#include <hip/hip_runtime.h>
#include <hip/hip_bf16.h>
#include <math.h>

// Fused graph-attention (cosine-sim softmax aggregation) for MI455X gfx1250.
//
// Roofline: adj (268 MB) read exactly once -> ~11.5us @ 23.3 TB/s HBM floor.
// 34 GFLOP of matmul hides under that stream via f16 WMMA. No NxN matrix is
// materialized; scores are bounded by beta<1 so softmax needs no running max.
//
// Key layout trick: the score GEMM computes S^T tiles, whose f32 C-layout
// coincides lane-for-lane with the f16 A-layout of the P@X GEMM -> the
// probability tile never leaves registers (no LDS transpose at all).

typedef __attribute__((ext_vector_type(16))) _Float16 v16h;
typedef __attribute__((ext_vector_type(8)))  _Float16 v8h;
typedef __attribute__((ext_vector_type(4)))  _Float16 v4h;
typedef __attribute__((ext_vector_type(8)))  float    v8f;
typedef __attribute__((ext_vector_type(4)))  float    v4f;

#define N_NODES 8192
#define D_FEAT  128

union V16U { v16h v; v8h h[2]; };

// ---------------------------------------------------------------------------
// Pre-pass: per-node reciprocal L2 norm, f16 row-major copy (xh) and f16
// feature-major copy (xhT). One wave per node; coalesced f32x4 loads.
// ---------------------------------------------------------------------------
__global__ __launch_bounds__(256)
void gat_prep_kernel(const float* __restrict__ x,
                     _Float16* __restrict__ xh,
                     _Float16* __restrict__ xhT,
                     float* __restrict__ rinv) {
    const int lane = threadIdx.x & 31;
    const int wave = threadIdx.x >> 5;
    const int node = blockIdx.x * 8 + wave;

    const float4 v = *(const float4*)(x + (size_t)node * D_FEAT + lane * 4);
    float ss = v.x * v.x + v.y * v.y + v.z * v.z + v.w * v.w;
    #pragma unroll
    for (int m = 1; m < 32; m <<= 1) ss += __shfl_xor(ss, m, 32);
    if (lane == 0) rinv[node] = rsqrtf(ss);     // 1/|x_node|

    v4h h;
    h[0] = (_Float16)v.x; h[1] = (_Float16)v.y;
    h[2] = (_Float16)v.z; h[3] = (_Float16)v.w;
    *(v4h*)(xh + (size_t)node * D_FEAT + lane * 4) = h;
    #pragma unroll
    for (int e = 0; e < 4; ++e)
        xhT[(size_t)(lane * 4 + e) * N_NODES + node] = h[e];
}

// ---------------------------------------------------------------------------
// Main fused kernel. One wave per 16-row output tile (512 tiles total).
// Streams adjacency columns in chunks of 32; 16 v_wmma per chunk, zero LDS.
// ---------------------------------------------------------------------------
__global__ __launch_bounds__(64)
void gat_main_kernel(const float* __restrict__ adj,
                     const float* __restrict__ beta_p,
                     const _Float16* __restrict__ xh,
                     const _Float16* __restrict__ xhT,
                     const float* __restrict__ rinv,
                     float* __restrict__ out) {
    const int lane = threadIdx.x & 31;
    const int wave = threadIdx.x >> 5;
    const int tile = blockIdx.x * 2 + wave;     // 0..511
    const int i0   = tile * 16;
    const int col  = lane & 15;
    const int hi   = lane >> 4;

    // B operand of S^T = X_cols @ X_rows^T : the i-rows, preloaded once.
    // B layout (16-bit, 32x16): lane n holds col n, K = {hi*8..+7, 16+hi*8..+7}.
    v16h Brow[4];
    #pragma unroll
    for (int kc = 0; kc < 4; ++kc) {
        V16U bu;
        const _Float16* rb = xh + (size_t)(i0 + col) * D_FEAT
                             + kc * 32 + hi * 8;
        bu.h[0] = *(const v8h*)(rb);
        bu.h[1] = *(const v8h*)(rb + 16);
        Brow[kc] = bu.v;
    }

    const float ascale = beta_p[0] * rinv[i0 + col];   // beta / |x_i|
    const float* adjrow = adj + (size_t)(i0 + col) * N_NODES;

    v8f acc[8];                                  // out tile, full D=128 in regs
    #pragma unroll
    for (int dt = 0; dt < 8; ++dt) acc[dt] = {};
    float zsum = 0.0f;                           // denominator for row i0+col

    for (int j0 = 0; j0 < N_NODES; j0 += 32) {
        // keep the HBM adj stream ahead of the matrix pipe
        __builtin_prefetch(adjrow + j0 + 1024, 0, 0);

        // ---- S^T tiles: A = X[j-chunk] (A layout: lane m = node j0+m,
        //      halves {0-7,16-23}/{8-15,24-31} by lane half) ----
        v8f s0 = {}, s1 = {};
        #pragma unroll
        for (int kc = 0; kc < 4; ++kc) {
            V16U a0u, a1u;
            const _Float16* r0 = xh + (size_t)(j0 + col) * D_FEAT
                                 + kc * 32 + hi * 8;
            a0u.h[0] = *(const v8h*)(r0);
            a0u.h[1] = *(const v8h*)(r0 + 16);
            s0 = __builtin_amdgcn_wmma_f32_16x16x32_f16(
                     false, a0u.v, false, Brow[kc], (short)0, s0, false, false);
            const _Float16* r1 = xh + (size_t)(j0 + 16 + col) * D_FEAT
                                 + kc * 32 + hi * 8;
            a1u.h[0] = *(const v8h*)(r1);
            a1u.h[1] = *(const v8h*)(r1 + 16);
            s1 = __builtin_amdgcn_wmma_f32_16x16x32_f16(
                     false, a1u.v, false, Brow[kc], (short)0, s1, false, false);
        }

        // ---- elementwise: p = adj * exp(beta * <xi,xj>/(|xi||xj|)).
        //      S^T C-layout == P A-layout, so pa is built entirely in-lane ----
        union { v16h v; _Float16 e[16]; } pa;
        #pragma unroll
        for (int t = 0; t < 2; ++t) {
            const int jb = j0 + t * 16 + hi * 8;          // 8 consecutive cols
            const v4f ad0 = __builtin_nontemporal_load((const v4f*)(adjrow + jb));
            const v4f ad1 = __builtin_nontemporal_load((const v4f*)(adjrow + jb + 4));
            const v4f c0 = *(const v4f*)(rinv + jb);
            const v4f c1 = *(const v4f*)(rinv + jb + 4);
            const v8f s = t ? s1 : s0;
            #pragma unroll
            for (int r = 0; r < 8; ++r) {
                const float aval = (r < 4) ? ad0[r] : ad1[r - 4];   // 0 or 1
                const float cj   = (r < 4) ? c0[r] : c1[r - 4];
                const float p    = aval * __expf(ascale * s[r] * cj);
                zsum += p;
                pa.e[t * 8 + r] = (_Float16)p;
            }
        }

        // ---- out += P @ X_cols : 8 tiles across D, B from feature-major ----
        #pragma unroll
        for (int dt = 0; dt < 8; ++dt) {
            V16U bxu;
            const _Float16* rx = xhT + (size_t)(dt * 16 + col) * N_NODES
                                 + j0 + hi * 8;
            bxu.h[0] = *(const v8h*)(rx);
            bxu.h[1] = *(const v8h*)(rx + 16);
            v16h Bx = bxu.v;
            acc[dt] = __builtin_amdgcn_wmma_f32_16x16x32_f16(
                          false, pa.v, false, Bx, (short)0, acc[dt],
                          false, false);
        }
    }

    // ---- finalize denominators: both lane-halves hold row i0+col ----
    const float z = zsum + __shfl_xor(zsum, 16, 32);
    const float zinv = 1.0f / z;                 // for row i0+col, all lanes

    // acc row r lives at global row i0+hi*8+r -> fetch its zinv from the lane
    // whose col equals that row index (value is half-symmetric).
    float zr[8];
    #pragma unroll
    for (int r = 0; r < 8; ++r)
        zr[r] = __shfl(zinv, hi * 8 + r, 32);

    // ---- normalize + store (each lane: 8 rows x 8 feature tiles) ----
    #pragma unroll
    for (int dt = 0; dt < 8; ++dt) {
        #pragma unroll
        for (int r = 0; r < 8; ++r) {
            __builtin_nontemporal_store(
                acc[dt][r] * zr[r],
                out + (size_t)(i0 + hi * 8 + r) * D_FEAT + dt * 16 + col);
        }
    }
}

// ---------------------------------------------------------------------------
extern "C" void kernel_launch(void* const* d_in, const int* in_sizes, int n_in,
                              void* d_out, int out_size, void* d_ws, size_t ws_size,
                              hipStream_t stream) {
    const float* x    = (const float*)d_in[0];   // [8192,128]
    const float* adj  = (const float*)d_in[1];   // [8192,8192]
    const float* beta = (const float*)d_in[2];   // [1]
    float* out = (float*)d_out;                  // [8192,128]

    // workspace carve-up: xh (2MB) | xhT (2MB) | rinv (32KB)
    _Float16* xh   = (_Float16*)d_ws;
    _Float16* xhT  = (_Float16*)((char*)d_ws + (2u << 20));
    float*    rinv = (float*)((char*)d_ws + (4u << 20));

    gat_prep_kernel<<<N_NODES / 8, 256, 0, stream>>>(x, xh, xhT, rinv);
    gat_main_kernel<<<512 / 2, 64, 0, stream>>>(adj, beta, xh, xhT, rinv, out);
}
